// MoE_50646254355321
// MI455X (gfx1250) — matrix operand order repaired and checked by reading, not compiled, not run on hardware
//
#include <hip/hip_runtime.h>
#include <hip/hip_bf16.h>
#include <stdint.h>

// ---------------- types ----------------
typedef __attribute__((ext_vector_type(16))) __bf16 v16bf;
typedef __attribute__((ext_vector_type(8)))  __bf16 v8bf;
typedef __attribute__((ext_vector_type(4)))  __bf16 v4bf;
typedef __attribute__((ext_vector_type(2)))  __bf16 v2bf;
typedef __attribute__((ext_vector_type(8)))  float  v8f;

#define N_TOK   1024
#define DIM     2048
#define F_DIM   1408
#define N_EXP   8

// float -> bf16, round-to-nearest-even (scalar fallback)
__device__ __forceinline__ __bf16 f2bf(float f) {
    union { float f; uint32_t u; } v; v.f = f;
    uint32_t r = v.u + 0x7FFFu + ((v.u >> 16) & 1u);
    uint16_t h = (uint16_t)(r >> 16);
    __bf16 out;
    __builtin_memcpy(&out, &h, 2);
    return out;
}

// packed pair convert: prefer HW v_cvt_pk_bf16_f32 when the builtin exists
__device__ __forceinline__ v2bf f2bf2(float a, float b) {
#if __has_builtin(__builtin_amdgcn_cvt_pk_bf16_f32)
    return __builtin_amdgcn_cvt_pk_bf16_f32(a, b);
#else
    v2bf r; r[0] = f2bf(a); r[1] = f2bf(b); return r;
#endif
}

// pack two float4 -> v8bf
__device__ __forceinline__ v8bf pack8(float4 a, float4 b) {
    v2bf p0 = f2bf2(a.x, a.y), p1 = f2bf2(a.z, a.w);
    v2bf p2 = f2bf2(b.x, b.y), p3 = f2bf2(b.z, b.w);
    v8bf r;
    r[0] = p0[0]; r[1] = p0[1]; r[2] = p1[0]; r[3] = p1[1];
    r[4] = p2[0]; r[5] = p2[1]; r[6] = p3[0]; r[7] = p3[1];
    return r;
}

// A-fragment (16x32 bf16, M x K) from a row-major bf16 row pointer.
// Contiguous K order (matches B): lane m (0-15): K = 0..15;
//          lane m+16:     K = 16..31.
__device__ __forceinline__ v16bf load_a_frag(const __bf16* rowptr, int khalf) {
    return *(const v16bf*)(rowptr + khalf * 16);
}

// B-fragment (32x16 bf16, K x N) from LDS tile laid out [col][k] with k-stride LDK.
// Contiguous K order (same as A): lanes 0-15 hold K=0..15, lanes 16-31 hold K=16..31.
template <int LDK>
__device__ __forceinline__ v16bf load_b_frag(const __bf16* lds, int col, int khalf) {
    return *(const v16bf*)(lds + col * LDK + khalf * 16);
}

// ---------------- small utility kernels ----------------
__global__ void init_cnt_kernel(int* cnt) {
    if (threadIdx.x < N_EXP) cnt[threadIdx.x] = 0;
}

__global__ void zero_out_kernel(float* out) {
    int idx = blockIdx.x * 256 + threadIdx.x;
    ((float4*)out)[idx] = make_float4(0.f, 0.f, 0.f, 0.f);
}

__global__ void cvt_x_kernel(const float* __restrict__ x, __bf16* __restrict__ xb) {
    int idx = blockIdx.x * 256 + threadIdx.x;   // one float4 per thread
    float4 v = ((const float4*)x)[idx];
    v2bf p0 = f2bf2(v.x, v.y), p1 = f2bf2(v.z, v.w);
    v4bf b; b[0] = p0[0]; b[1] = p0[1]; b[2] = p1[0]; b[3] = p1[1];
    ((v4bf*)xb)[idx] = b;
}

// ---------------- router: top-2 + softmax + per-expert compaction ----------------
__global__ __launch_bounds__(256) void router_kernel(
        const float* __restrict__ x, const float* __restrict__ gw,
        int* cnt, int* tokens, float* wts) {
    int wave = threadIdx.x >> 5, lane = threadIdx.x & 31;
    int tok = blockIdx.x * 8 + wave;
    const float* xr = x + (size_t)tok * DIM;
    float acc[N_EXP];
#pragma unroll
    for (int e = 0; e < N_EXP; ++e) acc[e] = 0.f;
    for (int d = lane; d < DIM; d += 32) {
        float xv = xr[d];
#pragma unroll
        for (int e = 0; e < N_EXP; ++e) acc[e] += xv * gw[e * DIM + d];
    }
#pragma unroll
    for (int e = 0; e < N_EXP; ++e)
#pragma unroll
        for (int off = 16; off > 0; off >>= 1)
            acc[e] += __shfl_xor(acc[e], off, 32);
    if (lane == 0) {
        int i0 = 0; float l0 = acc[0];
        for (int e = 1; e < N_EXP; ++e) if (acc[e] > l0) { l0 = acc[e]; i0 = e; }
        int i1 = -1; float l1 = -__builtin_inff();
        for (int e = 0; e < N_EXP; ++e)
            if (e != i0 && acc[e] > l1) { l1 = acc[e]; i1 = e; }
        float e1 = __expf(l1 - l0);
        float w0 = 1.0f / (1.0f + e1);
        float w1 = e1 * w0;
        int s0 = atomicAdd(&cnt[i0], 1);
        tokens[i0 * N_TOK + s0] = tok; wts[i0 * N_TOK + s0] = w0;
        int s1 = atomicAdd(&cnt[i1], 1);
        tokens[i1 * N_TOK + s1] = tok; wts[i1 * N_TOK + s1] = w1;
    }
}

// ---------------- FFN1: h = silu(x Wg^T) * (x Wu^T), bf16 WMMA ----------------
// Block: 256 threads (8 waves). Tile: 128 tokens x 32 F-cols. K staged 64 per sync.
__global__ __launch_bounds__(256) void ffn1_kernel(
        const float* __restrict__ gate_proj, const float* __restrict__ up_proj,
        const __bf16* __restrict__ xb, const int* __restrict__ cnt,
        const int* __restrict__ tokens, __bf16* __restrict__ h) {
    int e      = blockIdx.x >> 3;
    int tileM0 = (blockIdx.x & 7) * 128;
    int f0     = blockIdx.y * 32;
    int cntE   = cnt[e];
    if (tileM0 >= cntE) return;

    __shared__ __bf16 ldsBg[32 * 64];   // [col][k], 4 KB
    __shared__ __bf16 ldsBu[32 * 64];

    int t = threadIdx.x;
    int wave = t >> 5, lane = t & 31;
    int m_w = wave * 16;
    int rowL = lane & 15, khalf = lane >> 4;

    int slotA = tileM0 + m_w + rowL;
    int tokA  = (slotA < cntE) ? tokens[e * N_TOK + slotA] : 0;
    const __bf16* arow = xb + (size_t)tokA * DIM;

    int brow  = t >> 3;          // 32 rows, 8 threads each
    int bkoff = (t & 7) * 8;     // 8 floats per thread
    const float* gpr = gate_proj + ((size_t)e * F_DIM + f0 + brow) * DIM + bkoff;
    const float* upr = up_proj   + ((size_t)e * F_DIM + f0 + brow) * DIM + bkoff;

    v8f accG0 = {}, accG1 = {}, accU0 = {}, accU1 = {};

    for (int k0 = 0; k0 < DIM; k0 += 64) {
        float4 g4a = *(const float4*)(gpr + k0);
        float4 g4b = *(const float4*)(gpr + k0 + 4);
        float4 u4a = *(const float4*)(upr + k0);
        float4 u4b = *(const float4*)(upr + k0 + 4);
        __builtin_prefetch(gpr + k0 + 512, 0, 1);   // ~8 chunks ahead in weight stream
        __builtin_prefetch(upr + k0 + 512, 0, 1);
        *(v8bf*)(ldsBg + brow * 64 + bkoff) = pack8(g4a, g4b);
        *(v8bf*)(ldsBu + brow * 64 + bkoff) = pack8(u4a, u4b);
        __syncthreads();

#pragma unroll
        for (int s = 0; s < 2; ++s) {
            v16bf a   = load_a_frag(arow + k0 + s * 32, khalf);
            v16bf bg0 = load_b_frag<64>(ldsBg + s * 32, rowL,      khalf);
            v16bf bg1 = load_b_frag<64>(ldsBg + s * 32, 16 + rowL, khalf);
            v16bf bu0 = load_b_frag<64>(ldsBu + s * 32, rowL,      khalf);
            v16bf bu1 = load_b_frag<64>(ldsBu + s * 32, 16 + rowL, khalf);
            accG0 = __builtin_amdgcn_wmma_f32_16x16x32_bf16(false, a, false, bg0, (short)0, accG0, false, false);
            accG1 = __builtin_amdgcn_wmma_f32_16x16x32_bf16(false, a, false, bg1, (short)0, accG1, false, false);
            accU0 = __builtin_amdgcn_wmma_f32_16x16x32_bf16(false, a, false, bu0, (short)0, accU0, false, false);
            accU1 = __builtin_amdgcn_wmma_f32_16x16x32_bf16(false, a, false, bu1, (short)0, accU1, false, false);
        }
        __syncthreads();
    }

    // epilogue: SwiGLU (fast v_rcp sigmoid), store bf16 h
    int n = lane & 15, mb = khalf * 8;
#pragma unroll
    for (int sub = 0; sub < 2; ++sub) {
        v8f g = sub ? accG1 : accG0;
        v8f u = sub ? accU1 : accU0;
        int f = f0 + sub * 16 + n;
#pragma unroll
        for (int r = 0; r < 8; ++r) {
            int slot = tileM0 + m_w + mb + r;
            if (slot < cntE) {
                float gg = g[r];
                float sig = __builtin_amdgcn_rcpf(1.0f + __expf(-gg));
                float hh = gg * sig * u[r];
                h[((size_t)e * N_TOK + slot) * F_DIM + f] = f2bf(hh);
            }
        }
    }
}

// ---------------- FFN2: out += wt * (h Wd^T), bf16 WMMA + f32 atomic scatter ----------------
__global__ __launch_bounds__(256) void ffn2_kernel(
        const float* __restrict__ down_proj, const __bf16* __restrict__ h,
        const int* __restrict__ cnt, const int* __restrict__ tokens,
        const float* __restrict__ wts, float* __restrict__ out) {
    int e      = blockIdx.x >> 3;
    int tileM0 = (blockIdx.x & 7) * 128;
    int d0     = blockIdx.y * 32;
    int cntE   = cnt[e];
    if (tileM0 >= cntE) return;

    __shared__ __bf16 ldsB[32 * 64];    // [col][k], 4 KB

    int t = threadIdx.x;
    int wave = t >> 5, lane = t & 31;
    int m_w = wave * 16;
    int rowL = lane & 15, khalf = lane >> 4;

    const __bf16* arow = h + ((size_t)e * N_TOK + tileM0 + m_w + rowL) * F_DIM;

    int brow  = t >> 3;
    int bkoff = (t & 7) * 8;
    const float* dpr = down_proj + ((size_t)e * DIM + d0 + brow) * F_DIM + bkoff;

    v8f acc0 = {}, acc1 = {};

    for (int k0 = 0; k0 < F_DIM; k0 += 64) {
        float4 b4a = *(const float4*)(dpr + k0);
        float4 b4b = *(const float4*)(dpr + k0 + 4);
        __builtin_prefetch(dpr + k0 + 512, 0, 1);
        *(v8bf*)(ldsB + brow * 64 + bkoff) = pack8(b4a, b4b);
        __syncthreads();

#pragma unroll
        for (int s = 0; s < 2; ++s) {
            v16bf a  = load_a_frag(arow + k0 + s * 32, khalf);
            v16bf b0 = load_b_frag<64>(ldsB + s * 32, rowL,      khalf);
            v16bf b1 = load_b_frag<64>(ldsB + s * 32, 16 + rowL, khalf);
            acc0 = __builtin_amdgcn_wmma_f32_16x16x32_bf16(false, a, false, b0, (short)0, acc0, false, false);
            acc1 = __builtin_amdgcn_wmma_f32_16x16x32_bf16(false, a, false, b1, (short)0, acc1, false, false);
        }
        __syncthreads();
    }

    int n = lane & 15, mb = khalf * 8;
#pragma unroll
    for (int r = 0; r < 8; ++r) {
        int slot = tileM0 + m_w + mb + r;
        if (slot < cntE) {
            int   tok = tokens[e * N_TOK + slot];
            float wt  = wts[e * N_TOK + slot];
            atomicAdd(out + (size_t)tok * DIM + d0 + n,      acc0[r] * wt);
            atomicAdd(out + (size_t)tok * DIM + d0 + 16 + n, acc1[r] * wt);
        }
    }
}

// ---------------- launcher ----------------
extern "C" void kernel_launch(void* const* d_in, const int* in_sizes, int n_in,
                              void* d_out, int out_size, void* d_ws, size_t ws_size,
                              hipStream_t stream) {
    const float* x         = (const float*)d_in[0];
    const float* gate_w    = (const float*)d_in[1];
    const float* gate_proj = (const float*)d_in[2];
    const float* up_proj   = (const float*)d_in[3];
    const float* down_proj = (const float*)d_in[4];
    float* out = (float*)d_out;

    // workspace layout (bytes), ~27.3 MB total
    char* ws = (char*)d_ws;
    int*    cnt    = (int*)(ws + 0);                         // 8 ints
    int*    tokens = (int*)(ws + 256);                       // 8*1024 ints
    float*  wts    = (float*)(ws + 256 + 32768);             // 8*1024 floats
    __bf16* xb     = (__bf16*)(ws + 65792);                  // 1024*2048 bf16 = 4 MB
    __bf16* hbuf   = (__bf16*)(ws + 65792 + 4194304);        // 8*1024*1408 bf16 = 23 MB

    init_cnt_kernel<<<1, 32, 0, stream>>>(cnt);
    zero_out_kernel<<<(N_TOK * DIM) / (256 * 4), 256, 0, stream>>>(out);
    cvt_x_kernel<<<(N_TOK * DIM) / (256 * 4), 256, 0, stream>>>(x, xb);
    router_kernel<<<N_TOK / 8, 256, 0, stream>>>(x, gate_w, cnt, tokens, wts);
    ffn1_kernel<<<dim3(N_EXP * 8, F_DIM / 32), 256, 0, stream>>>(gate_proj, up_proj, xb, cnt, tokens, hbuf);
    ffn2_kernel<<<dim3(N_EXP * 8, DIM / 32), 256, 0, stream>>>(down_proj, hbuf, cnt, tokens, wts, out);
}
